// GNN_LSTM_Gravity_25838523253465
// MI455X (gfx1250) — matrix operand order repaired and checked by reading, not compiled, not run on hardware
//
#include <hip/hip_runtime.h>

// ---------------------------------------------------------------------------
// GNN (GCNConv) + single-step LSTM over edges + Linear(H,1), fused for gfx1250
// Dominant op: [E,128] x [128,192] edge-gate GEMM -> v_wmma_f32_16x16x32_bf16
// (f-gate skipped since c0 == 0). B matrix held entirely in VGPRs per wave.
// Activations use hardware v_tanh_f32 (CDNA5 TRANS op): 4 TRANS per h-element.
// ---------------------------------------------------------------------------

typedef __bf16  v16bf __attribute__((ext_vector_type(16)));
typedef float   v8f   __attribute__((ext_vector_type(8)));

union Frag16 { v16bf v; uint4 u[2]; };

__device__ __forceinline__ unsigned short f32_to_bf16(float f) {
  unsigned int u = __float_as_uint(f);
  unsigned int r = (u + 0x7FFFu + ((u >> 16) & 1u)) >> 16;   // RNE
  return (unsigned short)r;
}

// CDNA5 hardware tanh (transcendental VALU op, gfx1250)
__device__ __forceinline__ float htanh(float x) {
  float r;
  asm("v_tanh_f32 %0, %1" : "=v"(r) : "v"(x));
  return r;
}

// sigmoid(x) = 0.5*tanh(x/2) + 0.5  -> 1 TRANS + 2 FMA
__device__ __forceinline__ float hsigmoid(float x) {
  return __builtin_fmaf(0.5f, htanh(0.5f * x), 0.5f);
}

__device__ __forceinline__ void atomic_add_f32(float* p, float v) {
  __hip_atomic_fetch_add(p, v, __ATOMIC_RELAXED, __HIP_MEMORY_SCOPE_AGENT);
}

// ---------------- stage 0: zero agg[N*64] and deg[N] (contiguous) ----------
__global__ void k_zero(float* __restrict__ p, int n) {
  int i = blockIdx.x * blockDim.x + threadIdx.x;
  if (i < n) p[i] = 0.0f;
}

// ---------------- stage 1: xw = X @ W  ([N,16] x [16,64]) ------------------
__global__ void k_xw(const float* __restrict__ x, const float* __restrict__ w,
                     float* __restrict__ xw, int n64) {
  int idx = blockIdx.x * blockDim.x + threadIdx.x;
  if (idx >= n64) return;
  int i = idx >> 6, h = idx & 63;
  const float* xr = x + i * 16;
  float s = 0.f;
#pragma unroll
  for (int k = 0; k < 16; ++k) s += xr[k] * w[k * 64 + h];
  xw[idx] = s;
}

// ---------------- stage 2: degree scatter ----------------------------------
__global__ void k_deg(const long long* __restrict__ ecol, float* __restrict__ deg, int E) {
  int e = blockIdx.x * blockDim.x + threadIdx.x;
  if (e < E) atomic_add_f32(deg + ecol[e], 1.0f);
}

// ---------------- stage 3: dinv = rsqrt(deg + 1) ---------------------------
__global__ void k_dinv(const float* __restrict__ deg, float* __restrict__ dinv, int n) {
  int i = blockIdx.x * blockDim.x + threadIdx.x;
  if (i < n) dinv[i] = rsqrtf(deg[i] + 1.0f);
}

// ---------------- stage 4: agg[col] += norm * xw[row] ----------------------
__global__ void k_scatter(const long long* __restrict__ erow,
                          const long long* __restrict__ ecol,
                          const float* __restrict__ dinv,
                          const float* __restrict__ xw,
                          float* __restrict__ agg, int E) {
  int idx = blockIdx.x * blockDim.x + threadIdx.x;
  if (idx >= E * 16) return;
  int e = idx >> 4, q = (idx & 15) * 4;
  unsigned r = (unsigned)erow[e], c = (unsigned)ecol[e];
  float norm = dinv[r] * dinv[c];
  const float4 v = *(const float4*)(xw + (size_t)r * 64 + q);
  float* dst = agg + (size_t)c * 64 + q;
  atomic_add_f32(dst + 0, norm * v.x);
  atomic_add_f32(dst + 1, norm * v.y);
  atomic_add_f32(dst + 2, norm * v.z);
  atomic_add_f32(dst + 3, norm * v.w);
}

// ---------------- stage 5: emb = relu(agg + dinv^2*xw + bias) -> bf16 ------
__global__ void k_emb(const float* __restrict__ agg, const float* __restrict__ xw,
                      const float* __restrict__ dinv, const float* __restrict__ bias,
                      unsigned short* __restrict__ emb, int n64) {
  int idx = blockIdx.x * blockDim.x + threadIdx.x;
  if (idx >= n64) return;
  int i = idx >> 6, h = idx & 63;
  float d = dinv[i];
  float v = agg[idx] + d * d * xw[idx] + bias[h];
  emb[idx] = f32_to_bf16(v > 0.f ? v : 0.f);
}

// ---------------- stage 6: pack w_ih->bf16, bias256 = b_ih + b_hh ----------
__global__ void k_pack(const float* __restrict__ wih, const float* __restrict__ bih,
                       const float* __restrict__ bhh,
                       unsigned short* __restrict__ wihq, float* __restrict__ bias256) {
  int idx = blockIdx.x * blockDim.x + threadIdx.x;
  if (idx < 256 * 128) wihq[idx] = f32_to_bf16(wih[idx]);
  if (idx < 256)       bias256[idx] = bih[idx] + bhh[idx];
}

// ---------------- stage 7: fused edge-LSTM via WMMA ------------------------
// One wave processes 16 edges per tile, grid-striding over tiles. All 48
// B fragments (i/g/o gate columns of w_ih, bf16) live in VGPRs for the whole
// kernel; the inner loop is A-gather (global, L2-resident) + 48 WMMAs.
__global__ void __launch_bounds__(128, 1)
k_edge_lstm_wmma(const unsigned short* __restrict__ emb,   // [N,64] bf16
                 const unsigned short* __restrict__ wihq,  // [256,128] bf16 row-major
                 const float* __restrict__ bias256,        // b_ih + b_hh
                 const long long* __restrict__ erow,
                 const long long* __restrict__ ecol,
                 const float* __restrict__ fcw,            // [64]
                 const float* __restrict__ fcb,            // [1]
                 float* __restrict__ out, int E) {
  const int tid   = threadIdx.x;
  const int lane  = tid & 31;
  const int wave  = tid >> 5;            // 0..3
  const int m     = lane & 15;           // A row / C column index
  const int half8 = (lane >> 4) * 8;     // C row base

  const int nTiles = (E + 15) >> 4;
  const int nWaves = gridDim.x * 4;
  int tile = blockIdx.x * 4 + wave;      // wave-uniform
  if (tile >= nTiles) return;

  // ---- load all 48 B fragments into registers (one-time, L2 broadcast) ----
  // frag(nt,kt), lane ln, element e <-> w_ih[nt*16 + (ln&15)][kt*32 + (ln>>4)*16 + e]
  const int ntMap[12] = {0, 1, 2, 3, 8, 9, 10, 11, 12, 13, 14, 15};
  Frag16 B[12][4];
#pragma unroll
  for (int t = 0; t < 12; ++t) {
    const int row = ntMap[t] * 16 + m;
#pragma unroll
    for (int kt = 0; kt < 4; ++kt) {
      const unsigned short* src = wihq + row * 128 + kt * 32 + (lane >> 4) * 16;
      B[t][kt].u[0] = *(const uint4*)(src);
      B[t][kt].u[1] = *(const uint4*)(src + 8);
    }
  }

  // ---- per-lane constants: biases and fc weights at column nt*16 + m ------
  float bI[4], bG[4], bO[4], fw[4];
#pragma unroll
  for (int ht = 0; ht < 4; ++ht) {
    bI[ht] = bias256[      ht * 16 + m];   // i gate: cols   0..63
    bG[ht] = bias256[128 + ht * 16 + m];   // g gate: cols 128..191
    bO[ht] = bias256[192 + ht * 16 + m];   // o gate: cols 192..255
    fw[ht] = fcw[ht * 16 + m];
  }
  const float fb = fcb[0];

  // ---- A gather: ef row = [emb[row](64) | emb[col](64)], 16x32 bf16/K-tile.
  // lane<16: K = kt*32 + {0..15}; lane>=16: K = kt*32 + {16..31}
  // 32-bit element offsets -> SGPR-base + VGPR-offset (GVS) addressing.
  auto loadA = [&](int t, Frag16 A[4]) {
    int e = t * 16 + m; if (e >= E) e = E - 1;
    const unsigned r = (unsigned)erow[e];
    const unsigned c = (unsigned)ecol[e];
#pragma unroll
    for (int kt = 0; kt < 4; ++kt) {
      const unsigned node = (kt < 2) ? r : c;
      const unsigned off  = node * 64u + (unsigned)((kt & 1) * 32 + (lane >> 4) * 16);
      A[kt].u[0] = *(const uint4*)(emb + off);
      A[kt].u[1] = *(const uint4*)(emb + off + 8);
    }
  };

  Frag16 Acur[4], Anext[4];
  loadA(tile, Acur);

  while (true) {
    const int nxt = tile + nWaves;
    const bool hasNext = nxt < nTiles;       // wave-uniform
    if (hasNext) loadA(nxt, Anext);          // prefetch behind the WMMA chain

    // ---- 48 WMMAs: kt-outer so consecutive WMMAs reuse the A operand ------
    v8f acc[12];
#pragma unroll
    for (int t = 0; t < 12; ++t) acc[t] = (v8f){0.f,0.f,0.f,0.f,0.f,0.f,0.f,0.f};
#pragma unroll
    for (int kt = 0; kt < 4; ++kt) {
#pragma unroll
      for (int t = 0; t < 12; ++t) {
        acc[t] = __builtin_amdgcn_wmma_f32_16x16x32_bf16(
                     false, Acur[kt].v, false, B[t][kt].v, (short)0, acc[t],
                     false, false);
      }
    }

    // ---- LSTM elementwise + h . fc_w, all in the C-fragment layout --------
    // lane holds column n = ht*16 + m; VGPR j holds row half8 + j.
    float partial[8] = {0.f, 0.f, 0.f, 0.f, 0.f, 0.f, 0.f, 0.f};
#pragma unroll
    for (int ht = 0; ht < 4; ++ht) {
#pragma unroll
      for (int j = 0; j < 8; ++j) {
        float iv = acc[0 + ht][j] + bI[ht];
        float gv = acc[4 + ht][j] + bG[ht];
        float ov = acc[8 + ht][j] + bO[ht];
        float cv = hsigmoid(iv) * htanh(gv);
        float hv = hsigmoid(ov) * htanh(cv);
        partial[j] += hv * fw[ht];
      }
    }
    // reduce across the 16 columns (lanes) of each half
#pragma unroll
    for (int j = 0; j < 8; ++j) {
      float p = partial[j];
      p += __shfl_xor(p, 1);
      p += __shfl_xor(p, 2);
      p += __shfl_xor(p, 4);
      p += __shfl_xor(p, 8);
      partial[j] = p;
    }
    if (m == 0) {                 // lane 0 -> rows 0..7, lane 16 -> rows 8..15
      const int ob = tile * 16 + half8;
      if (ob + 8 <= E) {
        float4 s0 = {partial[0] + fb, partial[1] + fb, partial[2] + fb, partial[3] + fb};
        float4 s1 = {partial[4] + fb, partial[5] + fb, partial[6] + fb, partial[7] + fb};
        *(float4*)(out + ob)     = s0;
        *(float4*)(out + ob + 4) = s1;
      } else {
#pragma unroll
        for (int j = 0; j < 8; ++j)
          if (ob + j < E) out[ob + j] = partial[j] + fb;
      }
    }

    if (!hasNext) break;
    tile = nxt;
#pragma unroll
    for (int kt = 0; kt < 4; ++kt) {
      Acur[kt].u[0] = Anext[kt].u[0];
      Acur[kt].u[1] = Anext[kt].u[1];
    }
  }
}

// ---------------------------------------------------------------------------
extern "C" void kernel_launch(void* const* d_in, const int* in_sizes, int n_in,
                              void* d_out, int out_size, void* d_ws, size_t ws_size,
                              hipStream_t stream) {
  const float*     x    = (const float*)d_in[0];       // [N,16]
  const long long* eidx = (const long long*)d_in[1];   // [2,E] int64
  const float*     gw   = (const float*)d_in[2];       // [16,64]
  const float*     gb   = (const float*)d_in[3];       // [64]
  const float*     wih  = (const float*)d_in[4];       // [256,128]
  // d_in[5] = w_hh: unused (h0 == 0)
  const float*     bih  = (const float*)d_in[6];       // [256]
  const float*     bhh  = (const float*)d_in[7];       // [256]
  const float*     fcw  = (const float*)d_in[8];       // [64]
  const float*     fcb  = (const float*)d_in[9];       // [1]
  float*           out  = (float*)d_out;               // [E,1]

  const int N = in_sizes[0] / 16;
  const int E = in_sizes[1] / 2;
  const long long* erow = eidx;
  const long long* ecol = eidx + E;

  // workspace layout
  float*          agg   = (float*)d_ws;                  // N*64
  float*          deg   = agg + (size_t)N * 64;          // N   (contiguous w/ agg)
  float*          xw    = deg + N;                       // N*64
  float*          dinv  = xw + (size_t)N * 64;           // N
  unsigned short* emb   = (unsigned short*)(dinv + N);   // N*64 bf16
  unsigned short* wihq  = emb + (size_t)N * 64;          // 256*128 bf16
  float*          b256  = (float*)(wihq + 256 * 128);    // 256

  const int T = 256;
  auto cdiv = [](long long a, long long b) { return (int)((a + b - 1) / b); };

  k_zero<<<cdiv((long long)N * 65, T), T, 0, stream>>>(agg, N * 65);
  k_xw<<<cdiv((long long)N * 64, T), T, 0, stream>>>(x, gw, xw, N * 64);
  k_deg<<<cdiv(E, T), T, 0, stream>>>(ecol, deg, E);
  k_dinv<<<cdiv(N, T), T, 0, stream>>>(deg, dinv, N);
  k_scatter<<<cdiv((long long)E * 16, T), T, 0, stream>>>(erow, ecol, dinv, xw, agg, E);
  k_emb<<<cdiv((long long)N * 64, T), T, 0, stream>>>(agg, xw, dinv, gb, emb, N * 64);
  k_pack<<<cdiv(256 * 128, T), T, 0, stream>>>(wih, bih, bhh, wihq, b256);

  // 16 edges per wave-tile; 4 waves per 128-thread block; ~8 tiles per wave
  const int nTiles = (E + 15) / 16;
  const int blocks = cdiv(nTiles, 4 * 8);
  k_edge_lstm_wmma<<<blocks, 128, 0, stream>>>(emb, wihq, b256, erow, ecol,
                                               fcw, fcb, out, E);
}
